// ETFpreparer_60017872994897
// MI455X (gfx1250) — matrix operand order repaired and checked by reading, not compiled, not run on hardware
//
#include <hip/hip_runtime.h>
#include <stdint.h>

#define VOCAB 50257
#define DIM   768
#define BT    128          // macro tile (M and N)
#define KC    32           // K chunk per LDS stage (one bf16 WMMA K)
#define SPLITS 16          // split-K factor
#define NBLK  (DIM / BT)   // 6
#define NPAIR (NBLK * (NBLK + 1) / 2)  // 21 upper-triangular block pairs

typedef __bf16 v16bf __attribute__((ext_vector_type(16)));
typedef float  v8f   __attribute__((ext_vector_type(8)));

__device__ __forceinline__ unsigned short f2bf(float f) {
  __bf16 h = (__bf16)f;                       // native v_cvt on gfx1250, RNE
  return __builtin_bit_cast(unsigned short, h);
}
__device__ __forceinline__ float bf2f(unsigned short h) {
  return __uint_as_float(((uint32_t)h) << 16);
}

// ---------------- init: zero G and the two scalar accumulators ----------------
__global__ void init_kernel(float* __restrict__ G, float* __restrict__ acc) {
  int i = blockIdx.x * 256 + threadIdx.x;
  if (i < DIM * DIM) G[i] = 0.0f;
  if (i < 2) acc[i] = 0.0f;
}

// ------------- per-row RMS scale + diag^2 accumulation (one wave32/row) -------
__global__ __launch_bounds__(256) void rms_scales_kernel(
    const float* __restrict__ W, float* __restrict__ scales,
    float* __restrict__ diag_acc) {
  int row  = blockIdx.x * 8 + (threadIdx.x >> 5);
  int lane = threadIdx.x & 31;
  if (row >= VOCAB) return;
  const float* w = W + (size_t)row * DIM;
  float ssq = 0.0f;
  #pragma unroll
  for (int i = lane * 4; i < DIM; i += 32 * 4) {
    float4 v = *(const float4*)(w + i);
    ssq += v.x * v.x + v.y * v.y + v.z * v.z + v.w * v.w;
  }
  #pragma unroll
  for (int m = 16; m; m >>= 1) ssq += __shfl_xor(ssq, m, 32);
  if (lane == 0) {
    float ms = ssq * (1.0f / DIM);
    float s  = rsqrtf(ms + 1.1920929e-7f);   // float32 eps, as in F.rms_norm
    scales[row] = s;
    float dg = ssq * s * s;                  // ||wn_row||^2
    atomicAdd(diag_acc, dg * dg);
  }
}

// -------- fragment load: two aligned ds_load_b128 from swizzled LDS panel ----
__device__ __forceinline__ v16bf load_frag(const unsigned short* p, int col,
                                           int c0, int c1) {
  int sw = (col >> 2) & 3;
  int s0 = c0 ^ sw;
  int s1 = c1 ^ sw;
  union { v16bf v; uint4 q[2]; } u;
  u.q[0] = *(const uint4*)(p + col * KC + s0 * 8);
  u.q[1] = *(const uint4*)(p + col * KC + s1 * 8);
  return u.v;
}

// ---- register-stage prefetch of one K-chunk (npan panels) from global -------
__device__ __forceinline__ void stage_load(const float* __restrict__ W,
                                           const float* __restrict__ scales,
                                           int bm, int bn, int npan,
                                           int k0, int k_end, int tid,
                                           float4 rv[8], float rsc[8]) {
  #pragma unroll
  for (int p = 0; p < 2; ++p) {
    if (p >= npan) break;
    int cbase = (p == 0 ? bm : bn) * BT;
    #pragma unroll
    for (int i = 0; i < 4; ++i) {
      int idx = tid + 256 * i;          // 0..1023 float4 slots
      int k   = idx >> 5;               // 0..31
      int c4  = (idx & 31) << 2;        // column base (0..124)
      int kg  = k0 + k;
      float4 v = make_float4(0.f, 0.f, 0.f, 0.f);
      float  s = 0.f;
      if (kg < k_end) {
        v = *(const float4*)(W + (size_t)kg * DIM + cbase + c4);
        s = scales[kg];
      }
      rv[p * 4 + i]  = v;
      rsc[p * 4 + i] = s;
    }
  }
}

// ---- convert (scale, hi/lo bf16 split) and store registers into LDS --------
__device__ __forceinline__ void stage_store(unsigned short* __restrict__ pAh,
                                            unsigned short* __restrict__ pAl,
                                            unsigned short* __restrict__ pBh,
                                            unsigned short* __restrict__ pBl,
                                            int npan, int tid,
                                            const float4 rv[8],
                                            const float rsc[8]) {
  #pragma unroll
  for (int p = 0; p < 2; ++p) {
    if (p >= npan) break;
    unsigned short* ph = (p == 0) ? pAh : pBh;
    unsigned short* pl = (p == 0) ? pAl : pBl;
    #pragma unroll
    for (int i = 0; i < 4; ++i) {
      int idx = tid + 256 * i;
      int k   = idx >> 5;
      int c4  = (idx & 31) << 2;
      float4 v = rv[p * 4 + i];
      float  s = rsc[p * 4 + i];
      float vs[4] = {v.x * s, v.y * s, v.z * s, v.w * s};
      int chunk = k >> 3, kin = k & 7;
      #pragma unroll
      for (int e = 0; e < 4; ++e) {
        int col  = c4 + e;
        int slot = chunk ^ ((col >> 2) & 3);
        int addr = col * KC + slot * 8 + kin;
        unsigned short hb = f2bf(vs[e]);
        unsigned short lb = f2bf(vs[e] - bf2f(hb));
        ph[addr] = hb;
        pl[addr] = lb;
      }
    }
  }
}

// ---------------- main Gram kernel: G += Wn^T Wn via bf16x3 WMMA -------------
__global__ __launch_bounds__(256) void gram_kernel(
    const float* __restrict__ W, const float* __restrict__ scales,
    float* __restrict__ G) {
  // map blockIdx.x -> upper-triangular block pair (bm <= bn)
  int bm = 0, rem = blockIdx.x;
  while (rem >= NBLK - bm) { rem -= NBLK - bm; ++bm; }
  int bn = bm + rem;
  int npan = (bm == bn) ? 1 : 2;   // diagonal blocks: A-panel == B-panel

  const int KLEN    = (VOCAB + SPLITS - 1) / SPLITS;   // 3142
  int       k_begin = blockIdx.y * KLEN;
  int       k_end   = min(VOCAB, k_begin + KLEN);

  // panels stored column-major [col][k] with 16B-chunk XOR swizzle
  __shared__ unsigned short pAh[BT * KC];
  __shared__ unsigned short pAl[BT * KC];
  __shared__ unsigned short pBh[BT * KC];
  __shared__ unsigned short pBl[BT * KC];

  int tid  = threadIdx.x;
  int wave = tid >> 5;
  int lane = tid & 31;
  int lgrp = lane >> 4;    // 0 or 1
  int lrow = lane & 15;

  const unsigned short* Bh = (npan == 1) ? pAh : pBh;
  const unsigned short* Bl = (npan == 1) ? pAl : pBl;

  v8f Z = {0.f, 0.f, 0.f, 0.f, 0.f, 0.f, 0.f, 0.f};
  v8f C[8];
  #pragma unroll
  for (int t = 0; t < 8; ++t) C[t] = Z;

  float4 rv[8];
  float  rsc[8];
  stage_load(W, scales, bm, bn, npan, k_begin, k_end, tid, rv, rsc);

  for (int k0 = k_begin; k0 < k_end; k0 += KC) {
    __syncthreads();   // previous compute done reading LDS
    stage_store(pAh, pAl, pBh, pBl, npan, tid, rv, rsc);
    __syncthreads();

    // prefetch next chunk into registers; latency hides under the WMMAs below
    if (k0 + KC < k_end)
      stage_load(W, scales, bm, bn, npan, k0 + KC, k_end, tid, rv, rsc);

    // ---- compute: wave owns 16-row strip, 8 N-tiles, 3 WMMAs per tile ----
    int mcol = wave * 16 + lrow;                       // A-panel column
    v16bf Ah = load_frag(pAh, mcol, lgrp, lgrp + 2);   // K runs [8g,+8)∪[16+8g,+8)
    v16bf Al = load_frag(pAl, mcol, lgrp, lgrp + 2);
    #pragma unroll
    for (int t = 0; t < 8; ++t) {
      int ncol = t * 16 + lrow;                        // B-panel column
      v16bf Bhf = load_frag(Bh, ncol, lgrp, lgrp + 2);  // K [8g,+8)∪[16+8g,+8)
      v16bf Blf = load_frag(Bl, ncol, lgrp, lgrp + 2);
      C[t] = __builtin_amdgcn_wmma_f32_16x16x32_bf16(false, Ah, false, Bhf,
                                                     (short)0, C[t], false, false);
      C[t] = __builtin_amdgcn_wmma_f32_16x16x32_bf16(false, Ah, false, Blf,
                                                     (short)0, C[t], false, false);
      C[t] = __builtin_amdgcn_wmma_f32_16x16x32_bf16(false, Al, false, Bhf,
                                                     (short)0, C[t], false, false);
    }
  }

  // ---- accumulate split-K partials into G (fp32 atomics, L2-resident) ----
  int mrow0 = bm * BT + wave * 16 + 8 * lgrp;   // C layout: lanes16-31 -> M=r+8
  int ncol0 = bn * BT + lrow;
  #pragma unroll
  for (int t = 0; t < 8; ++t) {
    #pragma unroll
    for (int r = 0; r < 8; ++r) {
      atomicAdd(&G[(size_t)(mrow0 + r) * DIM + ncol0 + t * 16], C[t][r]);
    }
  }
}

// --------- Frobenius^2 over computed blocks (off-diagonal weighted 2x) -------
__global__ __launch_bounds__(256) void sqsum_kernel(const float* __restrict__ G,
                                                    float* __restrict__ acc) {
  int bm = 0, rem = blockIdx.x;
  while (rem >= NBLK - bm) { rem -= NBLK - bm; ++bm; }
  int bn = bm + rem;
  float wgt = (bm == bn) ? 1.0f : 2.0f;
  float s = 0.0f;
  for (int i = threadIdx.x; i < BT * BT; i += 256) {
    int r = i >> 7, c = i & (BT - 1);
    float g = G[(size_t)(bm * BT + r) * DIM + bn * BT + c];
    s += g * g;
  }
  #pragma unroll
  for (int m = 16; m; m >>= 1) s += __shfl_xor(s, m, 32);
  __shared__ float red[8];
  if ((threadIdx.x & 31) == 0) red[threadIdx.x >> 5] = s;
  __syncthreads();
  if (threadIdx.x == 0) {
    float t = 0.f;
    #pragma unroll
    for (int i = 0; i < 8; ++i) t += red[i];
    atomicAdd(&acc[0], t * wgt);
  }
}

// ------------------------------ finalize scalar ------------------------------
__global__ void finalize_kernel(const float* __restrict__ acc,
                                float* __restrict__ out) {
  double total = (double)acc[0];
  double diag  = (double)acc[1];
  double denom = (double)VOCAB * (double)(VOCAB - 1);
  out[0] = (float)((total - diag) / denom);
}

extern "C" void kernel_launch(void* const* d_in, const int* in_sizes, int n_in,
                              void* d_out, int out_size, void* d_ws, size_t ws_size,
                              hipStream_t stream) {
  (void)in_sizes; (void)n_in; (void)out_size; (void)ws_size;
  const float* W = (const float*)d_in[0];   // [VOCAB, DIM] fp32

  float* ws     = (float*)d_ws;
  float* scales = ws;                                   // VOCAB floats
  float* G      = ws + ((VOCAB + 255) / 256) * 256;     // 768*768 floats
  float* acc    = G + DIM * DIM;                        // [0]=||G||^2, [1]=diag^2

  init_kernel<<<(DIM * DIM + 255) / 256, 256, 0, stream>>>(G, acc);
  rms_scales_kernel<<<(VOCAB + 7) / 8, 256, 0, stream>>>(W, scales, acc + 1);

  dim3 grid(NPAIR, SPLITS);
  gram_kernel<<<grid, 256, 0, stream>>>(W, scales, G);

  sqsum_kernel<<<NPAIR, 256, 0, stream>>>(G, acc);
  finalize_kernel<<<1, 1, 0, stream>>>(acc, (float*)d_out);
}
